// GraphAttentionLayer_27960237097331
// MI455X (gfx1250) — matrix operand order repaired and checked by reading, not compiled, not run on hardware
//
#include <hip/hip_runtime.h>
#include <cstdint>

typedef __attribute__((ext_vector_type(16))) _Float16 v16h;
typedef __attribute__((ext_vector_type(8)))  _Float16 v8h;
typedef __attribute__((ext_vector_type(8)))  float    v8f;

namespace {
constexpr int kBS = 8, kNV = 16, kMF = 16, kMT = 16, kNP = 8, kKK = 3, kNR = 12;
constexpr int kH = 64, kNOUT = 64;
constexpr int kROWS   = kMT * kMF;        // 256 (t,f) rows per (b,v)
constexpr int kRelRow = kNP * kKK * kNR;  // 288 floats per (f,t)
constexpr int kRelV4  = kRelRow / 4;      // 72 float4 per row
constexpr int kZP = 72;                   // z row pitch in halfs (144B, 16B aligned)
constexpr int kWP = 72;                   // out_w row pitch in halfs
}

__global__ __launch_bounds__(256)
void gat_fused_kernel(const float* __restrict__ rel,      // [BS,NV,MF,MT,NP,KK,NR]
                      const float* __restrict__ src,      // [BS,NV,MT,MF,H]
                      const float* __restrict__ svec,     // [BS,NV,NOUT]
                      const float* __restrict__ wstack,   // [NR,H,H]
                      const float* __restrict__ lin_w,    // [H+NOUT]
                      const float* __restrict__ lin_b,    // [1]
                      const float* __restrict__ out_w,    // [NOUT,H]
                      const float* __restrict__ out_b,    // [NOUT]
                      float* __restrict__ dst,            // out or partials
                      int wpb)                            // WGs per (b,v): 1 or 4
{
  __shared__ float    diag_s[kNR][kH];
  __shared__ float    wz_s[kH];
  __shared__ _Float16 w16_s[kNOUT][kWP];
  __shared__ _Float16 z_s[kROWS][kZP];
  __shared__ float4   relbuf[8][2][kRelV4];   // per-wave double buffer
  __shared__ float    acc_s[kNOUT];
  __shared__ float    sdot_s;

  const int bv   = blockIdx.x / wpb;
  const int rq   = blockIdx.x - bv * wpb;
  const int rows = kROWS / wpb;
  const int row0 = rq * rows;

  const int tid  = threadIdx.x;
  const int lane = tid & 31;
  const int wave = tid >> 5;

  // ---------------- stage shared constants ----------------
  for (int i = tid; i < kNR * kH; i += 256) {
    int r = i >> 6, d = i & 63;
    diag_s[r][d] = wstack[(r * kH + d) * kH + d];      // diag of Wstack[r]
  }
  for (int i = tid; i < kH; i += 256) wz_s[i] = lin_w[kNOUT + i];
  for (int i = tid; i < kNOUT * kH; i += 256) {
    int o = i >> 6, d = i & 63;
    w16_s[o][d] = (_Float16)out_w[o * kH + d];
  }
  if (tid < kNOUT) acc_s[tid] = 0.0f;
  if (wave == 0) {  // s·w_s + lin_b (scalar per (b,v))
    float part = svec[bv * kNOUT + lane]      * lin_w[lane]
               + svec[bv * kNOUT + 32 + lane] * lin_w[32 + lane];
    #pragma unroll
    for (int off = 16; off > 0; off >>= 1) part += __shfl_xor(part, off, 32);
    if (lane == 0) sdot_s = part + lin_b[0];
  }
  __syncthreads();

  const float sdot = sdot_s;
  const int d0 = 2 * lane, d1 = 2 * lane + 1;   // each lane owns 2 of H=64 dims
  float dg0[kNR], dg1[kNR];
  #pragma unroll
  for (int r = 0; r < kNR; ++r) { dg0[r] = diag_s[r][d0]; dg1[r] = diag_s[r][d1]; }
  const float wz0 = wz_s[d0], wz1 = wz_s[d1];

  const float* relbv = rel + (size_t)bv * kMF * kMT * kRelRow;
  const float* srcbv = src + (size_t)bv * kMT * kMF * kH;

  // async cache->LDS copy of one 288-float rel slab (3 instructions per wave,
  // ASYNCcnt-tracked, no VGPR round trip)
  auto issue_batch = [&](int rid, int buf) {
    const int t = rid >> 4, f = rid & 15;
    const float4* g = (const float4*)(relbv + (size_t)(f * kMT + t) * kRelRow);
    uint32_t la0 = (uint32_t)(uintptr_t)&relbuf[wave][buf][lane];
    uint64_t ga0 = (uint64_t)(uintptr_t)(g + lane);
    asm volatile("global_load_async_to_lds_b128 %0, %1, off"
                 :: "v"(la0), "v"(ga0) : "memory");
    uint32_t la1 = (uint32_t)(uintptr_t)&relbuf[wave][buf][lane + 32];
    uint64_t ga1 = (uint64_t)(uintptr_t)(g + lane + 32);
    asm volatile("global_load_async_to_lds_b128 %0, %1, off"
                 :: "v"(la1), "v"(ga1) : "memory");
    if (lane < 8) {  // tail: 72 = 32+32+8 (EXEC-masked, still 1 instr/wave)
      uint32_t la2 = (uint32_t)(uintptr_t)&relbuf[wave][buf][lane + 64];
      uint64_t ga2 = (uint64_t)(uintptr_t)(g + lane + 64);
      asm volatile("global_load_async_to_lds_b128 %0, %1, off"
                   :: "v"(la2), "v"(ga2) : "memory");
    }
  };

  // ---------------- per-row fused attention (pipelined) ----------------
  int cur = 0;
  if (row0 + wave < row0 + rows) issue_batch(row0 + wave, cur);
  for (int rowid = row0 + wave; rowid < row0 + rows; rowid += 8) {
    const int t = rowid >> 4, f = rowid & 15;
    const bool have_next = (rowid + 8 < row0 + rows);
    if (have_next) {
      issue_batch(rowid + 8, cur ^ 1);     // overlap next slab with compute
      if (rowid + 16 < row0 + rows) {      // L2 stream-ahead hint, 2 rows out
        int nr = rowid + 16, nt = nr >> 4, nf = nr & 15;
        __builtin_prefetch(relbv + (size_t)(nf * kMT + nt) * kRelRow + 4 * lane, 0, 3);
      }
      // async loads complete in order: cnt<=3 => current row's 3 loads landed
      asm volatile("s_wait_asynccnt 0x3" ::: "memory");
    } else {
      asm volatile("s_wait_asynccnt 0x0" ::: "memory");
    }
    const float* myrel = (const float*)relbuf[wave][cur];

    const float2 sv = *(const float2*)(srcbv + (size_t)(t * kMF + f) * kH + d0);

    float zj0[kNP], zj1[kNP], lg[kNP];
    #pragma unroll
    for (int p = 0; p < kNP; ++p) {
      float w0 = 1.0f, w1 = 1.0f;
      #pragma unroll
      for (int k = 0; k < kKK; ++k) {
        const float* rr = myrel + (p * kKK + k) * kNR;   // wave-uniform (broadcast)
        float a0 = 0.0f, a1 = 0.0f;
        #pragma unroll
        for (int r = 0; r < kNR; ++r) {
          const float rv = rr[r];
          a0 = fmaf(rv, dg0[r], a0);
          a1 = fmaf(rv, dg1[r], a1);
        }
        w0 *= a0; w1 *= a1;
      }
      const float z0 = w0 * sv.x, z1 = w1 * sv.y;
      zj0[p] = z0; zj1[p] = z1;
      float l = z0 * wz0 + z1 * wz1;
      #pragma unroll
      for (int off = 16; off > 0; off >>= 1) l += __shfl_xor(l, off, 32);
      lg[p] = l + sdot;
    }
    // softmax over NP=8 (replicated across lanes, all-f32)
    float m = lg[0];
    #pragma unroll
    for (int p = 1; p < kNP; ++p) m = fmaxf(m, lg[p]);
    float den = 0.0f;
    #pragma unroll
    for (int p = 0; p < kNP; ++p) { lg[p] = __expf(lg[p] - m); den += lg[p]; }
    const float inv = 1.0f / den;
    float za = 0.0f, zb = 0.0f;
    #pragma unroll
    for (int p = 0; p < kNP; ++p) {
      const float a = lg[p] * inv;
      za = fmaf(a, zj0[p], za);
      zb = fmaf(a, zj1[p], zb);
    }
    union { _Float16 h[2]; unsigned u; } pk;
    pk.h[0] = (_Float16)za; pk.h[1] = (_Float16)zb;
    *(unsigned*)&z_s[rowid][d0] = pk.u;   // packed f16 store, conflict-free
    cur ^= 1;
  }
  __syncthreads();

  // ------- projection: relu(z @ out_w^T + out_b), row-reduced, via WMMA -------
  const int ntiles = (rows >> 4) * 4;     // (row tiles) x (4 col tiles of N=64)
  for (int tl = wave; tl < ntiles; tl += 8) {
    const int rt   = (row0 >> 4) + (tl >> 2);
    const int ct   = tl & 3;
    const int row  = rt * 16 + (lane & 15);
    const int hi   = lane >> 4;
    const int ocol = ct * 16 + (lane & 15);
    v8f c = {};
    #pragma unroll
    for (int kb = 0; kb < 2; ++kb) {      // K=64 = 2 x 32
      // A (16x32 f16, ISA lane layout): lane<16 -> K 0..7,16..23 ; lane>=16 -> +8
      const v8h alo = *(const v8h*)&z_s[row][kb * 32 + hi * 8];
      const v8h ahi = *(const v8h*)&z_s[row][kb * 32 + 16 + hi * 8];
      const v16h a = __builtin_shufflevector(alo, ahi,
          0,1,2,3,4,5,6,7,8,9,10,11,12,13,14,15);
      // B (32x16 f16): lane holds N=lane%16, K same as A: 0..7,16..23 ; lane>=16 -> +8 ; B[K][N]=out_w[N][d]
      const v8h blo = *(const v8h*)&w16_s[ocol][kb * 32 + hi * 8];
      const v8h bhi = *(const v8h*)&w16_s[ocol][kb * 32 + 16 + hi * 8];
      const v16h b = __builtin_shufflevector(blo, bhi,
          0,1,2,3,4,5,6,7,8,9,10,11,12,13,14,15);
      c = __builtin_amdgcn_wmma_f32_16x16x32_f16(false, a, false, b,
                                                 (short)0, c, false, false);
    }
    // D layout: lane holds N=ocol, rows M = (lane/16)*8 + i  -> bias, relu, row-sum
    const float ob = out_b[ocol];
    float sum = 0.0f;
    #pragma unroll
    for (int i = 0; i < 8; ++i) sum += fmaxf(c[i] + ob, 0.0f);
    sum += __shfl_xor(sum, 16, 32);       // combine the two M-halves (same N)
    if (hi == 0) atomicAdd(&acc_s[ocol], sum);   // ds_add_f32
  }
  __syncthreads();
  if (tid < kNOUT) dst[(size_t)blockIdx.x * kNOUT + tid] = acc_s[tid];
}

__global__ __launch_bounds__(256)
void gat_reduce_kernel(const float* __restrict__ part, float* __restrict__ out) {
  const int i = blockIdx.x * blockDim.x + threadIdx.x;   // 8192 outputs
  if (i < kBS * kNV * kNOUT) {
    const int bv = i >> 6, o = i & 63;
    const float* p = part + (size_t)bv * 4 * kNOUT + o;
    out[i] = p[0] + p[kNOUT] + p[2 * kNOUT] + p[3 * kNOUT];
  }
}

extern "C" void kernel_launch(void* const* d_in, const int* in_sizes, int n_in,
                              void* d_out, int out_size, void* d_ws, size_t ws_size,
                              hipStream_t stream) {
  const float* rel   = (const float*)d_in[0];
  const float* src   = (const float*)d_in[1];
  const float* svec  = (const float*)d_in[2];
  const float* wst   = (const float*)d_in[3];
  const float* lin_w = (const float*)d_in[4];
  const float* lin_b = (const float*)d_in[5];
  const float* out_w = (const float*)d_in[6];
  const float* out_b = (const float*)d_in[7];
  float* outp = (float*)d_out;

  const size_t part_bytes = (size_t)kBS * kNV * 4 * kNOUT * sizeof(float);
  const int wpb = (d_ws != nullptr && ws_size >= part_bytes) ? 4 : 1;
  float* dst = (wpb == 4) ? (float*)d_ws : outp;

  dim3 grid(kBS * kNV * wpb), block(256);
  gat_fused_kernel<<<grid, block, 0, stream>>>(rel, src, svec, wst, lin_w, lin_b,
                                               out_w, out_b, dst, wpb);
  if (wpb == 4) {
    const int n = kBS * kNV * kNOUT;
    gat_reduce_kernel<<<(n + 255) / 256, 256, 0, stream>>>((const float*)d_ws, outp);
  }
}
